// EdgePredictorNetwork_17059610100214
// MI455X (gfx1250) — matrix operand order repaired and checked by reading, not compiled, not run on hardware
//
#include <hip/hip_runtime.h>
#include <hip/hip_bf16.h>
#include <cstdint>

#define NNODES 1024
#define NF 27
#define EF 17
#define H 128
#define HD 32
#define HEADS 4

typedef __attribute__((ext_vector_type(16))) __bf16 v16bf;
typedef __attribute__((ext_vector_type(8)))  float  v8f;

// ---------------- Kernel 1: node encoder (relu(X@W1+b1)@W2+b2) ----------------
__global__ __launch_bounds__(128) void enc_kernel(
    const float* __restrict__ x,
    const float* __restrict__ w1, const float* __restrict__ b1,
    const float* __restrict__ w2, const float* __restrict__ b2,
    float* __restrict__ enc)
{
    __shared__ float xr[32];
    __shared__ float h1[H];
    int i = blockIdx.x;
    int t = threadIdx.x;
    if (t < NF) xr[t] = x[i * NF + t];
    __syncthreads();
    float a = b1[t];
    #pragma unroll
    for (int c = 0; c < NF; ++c) a += xr[c] * w1[c * H + t];
    h1[t] = fmaxf(a, 0.f);
    __syncthreads();
    float o = b2[t];
    for (int c = 0; c < H; ++c) o += h1[c] * w2[c * H + t];
    enc[(size_t)i * H + t] = o;
}

// ---------------- Kernel 2: all linear projections off enc ----------------
__global__ __launch_bounds__(128) void proj_kernel(
    const float* __restrict__ enc,
    const float* __restrict__ ipw, const float* __restrict__ ipb,
    const float* __restrict__ pw1,
    const float* __restrict__ cw1,
    float* __restrict__ q, float* __restrict__ k,
    float* __restrict__ sap, float* __restrict__ sbp,
    float* __restrict__ sac, float* __restrict__ sbc)
{
    __shared__ float e[H];
    int i = blockIdx.x, t = threadIdx.x;
    e[t] = enc[(size_t)i * H + t];
    __syncthreads();
    float accq = ipb[t], acck = ipb[H + t], ap = 0.f, bp = 0.f;
    for (int c = 0; c < H; ++c) {
        float ec = e[c];
        accq += ec * ipw[c * 3 * H + t];
        acck += ec * ipw[c * 3 * H + H + t];
        ap   += ec * pw1[c * H + t];
        bp   += ec * pw1[(H + c) * H + t];
    }
    q[(size_t)i * H + t] = accq;
    k[(size_t)i * H + t] = acck;
    sap[(size_t)i * H + t] = ap;
    sbp[(size_t)i * H + t] = bp;
    if (t < H / 2) {
        float ac = 0.f, bc = 0.f;
        for (int c = 0; c < H; ++c) {
            float ec = e[c];
            ac += ec * cw1[c * (H / 2) + t];
            bc += ec * cw1[(H + c) * (H / 2) + t];
        }
        sac[(size_t)i * (H / 2) + t] = ac;
        sbc[(size_t)i * (H / 2) + t] = bc;
    }
}

// ---------------- Kernel 3: attention weights (softmax over k, head mean) ----------------
__global__ __launch_bounds__(128) void attn_kernel(
    const float* __restrict__ q, const float* __restrict__ k,
    float* __restrict__ out)
{
    __shared__ float sc[HEADS][NNODES];   // 16 KB
    __shared__ float inv[HEADS];
    int i = blockIdx.x;
    int lane = threadIdx.x & 31;
    int h = threadIdx.x >> 5;            // one wave per head
    float qv[HD];
    const float* qp = q + (size_t)i * H + h * HD;
    #pragma unroll
    for (int t = 0; t < HD; ++t) qv[t] = qp[t];
    const float scale = 0.17677669529663687f; // 1/sqrt(32)
    float mx = -3.4e38f;
    for (int c = 0; c < NNODES / 32; ++c) {
        int j = c * 32 + lane;
        const float* kp = k + (size_t)j * H + h * HD;
        float d = 0.f;
        #pragma unroll
        for (int t = 0; t < HD; ++t) d += qv[t] * kp[t];
        d *= scale;
        sc[h][j] = d;
        mx = fmaxf(mx, d);
    }
    #pragma unroll
    for (int o = 16; o > 0; o >>= 1) mx = fmaxf(mx, __shfl_xor(mx, o, 32));
    float s = 0.f;
    for (int c = 0; c < NNODES / 32; ++c) {
        int j = c * 32 + lane;
        float ev = __expf(sc[h][j] - mx);
        sc[h][j] = ev;
        s += ev;
    }
    #pragma unroll
    for (int o = 16; o > 0; o >>= 1) s += __shfl_xor(s, o, 32);
    if (lane == 0) inv[h] = 1.f / s;
    __syncthreads();
    for (int j = threadIdx.x; j < NNODES; j += 128) {
        out[(size_t)i * NNODES + j] = 0.25f *
            (sc[0][j] * inv[0] + sc[1][j] * inv[1] + sc[2][j] * inv[2] + sc[3][j] * inv[3]);
    }
}

// ---------------- Kernel 4: edge features (WMMA bf16) + edge probabilities ----------------
// Per block: one source node i. 8 waves, each handles 8 tiles of 16 target nodes j.
// A tile: hp[16 j x 32 K] built in-register in the 16-bit A layout (lane m=L%16,
// K-halves by L/16). B: pred_w2 (128x17 -> padded 128x32) staged to LDS transposed
// as bf16 and held in registers for the whole j loop.
__global__ __launch_bounds__(256) void edge_kernel(
    const float* __restrict__ sap, const float* __restrict__ sbp,
    const float* __restrict__ sac, const float* __restrict__ sbc,
    const float* __restrict__ pb1, const float* __restrict__ w2, const float* __restrict__ pb2,
    const float* __restrict__ cb1, const float* __restrict__ cw2, const float* __restrict__ cb2,
    float* __restrict__ out_prob, float* __restrict__ out_ef)
{
    __shared__ alignas(16) float app[H];              // sa_p[i,:] + pred_b1
    __shared__ alignas(16) __bf16 w2t[32][H + 8];     // bf16, [col][k], padded
    __shared__ float sacb[H / 2];                     // sa_c[i,:] + cls_b1
    __shared__ float w2c[H / 2];
    __shared__ float pb2s[EF];

    int i = blockIdx.x;
    int tid = threadIdx.x;

    if (tid < H) app[tid] = sap[(size_t)i * H + tid] + pb1[tid];
    if (tid < H / 2) { sacb[tid] = sac[(size_t)i * (H / 2) + tid] + cb1[tid]; w2c[tid] = cw2[tid]; }
    if (tid < EF) pb2s[tid] = pb2[tid];
    for (int idx = tid; idx < 32 * H; idx += 256) {
        int c = idx >> 7, kk = idx & (H - 1);
        float v = (c < EF) ? w2[kk * EF + c] : 0.f;
        w2t[c][kk] = (__bf16)v;
    }
    __syncthreads();

    int lane = tid & 31;
    int wv = tid >> 5;        // wave 0..7
    int m = lane & 15;        // A row (j within tile) / B,C,D column
    int half = lane >> 4;     // lane half selects K range
    int abase = half * 8;     // A-layout K base within 32-K slab
    int bbase = half * 16;    // unused: B is loaded from abase too

    // B operand registers: 4 K-slabs x 2 N-tiles (cols 0..15 and 16..31)
    union BV { v16bf v; uint4 qd[2]; };
    BV B0[4], B1[4];
    #pragma unroll
    for (int s = 0; s < 4; ++s) {
        int k0 = s * 32;
        const uint4* p0 = (const uint4*)&w2t[m][k0 + abase];
        B0[s].qd[0] = p0[0]; B0[s].qd[1] = p0[2];
        const uint4* p1 = (const uint4*)&w2t[16 + m][k0 + abase];
        B1[s].qd[0] = p1[0]; B1[s].qd[1] = p1[2];
    }

    // Lane-local copy of sa_p+b terms in A order (reused for all 64 tiles)
    float apv[4][16];
    #pragma unroll
    for (int s = 0; s < 4; ++s) {
        int k0 = s * 32;
        #pragma unroll
        for (int e = 0; e < 16; ++e) {
            int kk = k0 + abase + e + ((e >= 8) ? 8 : 0);
            apv[s][e] = app[kk];
        }
    }
    float cb2v = cb2[0];

    for (int t = 0; t < 8; ++t) {
        int j0 = (wv * 8 + t) * 16;
        int j = j0 + m;
        const float* sbrow = sbp + (size_t)j * H;
        v8f acc0 = {}; v8f acc1 = {};
        #pragma unroll
        for (int s = 0; s < 4; ++s) {
            int k0 = s * 32;
            v16bf A;
            #pragma unroll
            for (int e = 0; e < 16; ++e) {
                int kk = k0 + abase + e + ((e >= 8) ? 8 : 0);
                float hv = fmaxf(apv[s][e] + sbrow[kk], 0.f);
                A[e] = (__bf16)hv;   // native f32->bf16 convert (packed cvt)
            }
            acc0 = __builtin_amdgcn_wmma_f32_16x16x32_bf16(false, A, false, B0[s].v,
                                                           (short)0, acc0, false, false);
            acc1 = __builtin_amdgcn_wmma_f32_16x16x32_bf16(false, A, false, B1[s].v,
                                                           (short)0, acc1, false, false);
        }
        // Edge classifier: lane half sums 32 of 64 K, combine across halves
        const float* sbcrow = sbc + (size_t)j * (H / 2);
        int kb = half * 32;
        float part = 0.f;
        #pragma unroll
        for (int kk = 0; kk < 32; ++kk) {
            float hv = sacb[kb + kk] + sbcrow[kb + kk];
            part += fmaxf(hv, 0.f) * w2c[kb + kk];
        }
        float other = __shfl_xor(part, 16, 32);
        if (half == 0) {
            float z = part + other + cb2v;
            out_prob[(size_t)i * NNODES + j] = 1.f / (1.f + __expf(-z));
        }
        // Store D: element (M = v + 8*half, N = m); tile1 only column e=16 is real
        #pragma unroll
        for (int v = 0; v < 8; ++v) {
            int jj = j0 + v + half * 8;
            float* orow = out_ef + ((size_t)i * NNODES + jj) * EF;
            orow[m] = acc0[v] + pb2s[m];
            if (m == 0) orow[16] = acc1[v] + pb2s[16];
        }
    }
}

extern "C" void kernel_launch(void* const* d_in, const int* in_sizes, int n_in,
                              void* d_out, int out_size, void* d_ws, size_t ws_size,
                              hipStream_t stream) {
    const float* x      = (const float*)d_in[0];
    const float* enc_w1 = (const float*)d_in[1];
    const float* enc_b1 = (const float*)d_in[2];
    const float* enc_w2 = (const float*)d_in[3];
    const float* enc_b2 = (const float*)d_in[4];
    const float* ipw    = (const float*)d_in[5];
    const float* ipb    = (const float*)d_in[6];
    const float* pw1    = (const float*)d_in[7];
    const float* pb1    = (const float*)d_in[8];
    const float* pw2    = (const float*)d_in[9];
    const float* pb2    = (const float*)d_in[10];
    const float* cw1    = (const float*)d_in[11];
    const float* cb1    = (const float*)d_in[12];
    const float* cw2    = (const float*)d_in[13];
    const float* cb2    = (const float*)d_in[14];

    float* ws  = (float*)d_ws;
    float* enc = ws;
    float* q   = enc + (size_t)NNODES * H;
    float* k   = q   + (size_t)NNODES * H;
    float* sap = k   + (size_t)NNODES * H;
    float* sbp = sap + (size_t)NNODES * H;
    float* sac = sbp + (size_t)NNODES * H;
    float* sbc = sac + (size_t)NNODES * (H / 2);

    float* out      = (float*)d_out;
    float* out_attn = out;
    float* out_prob = out + (size_t)NNODES * NNODES;
    float* out_ef   = out + 2 * (size_t)NNODES * NNODES;

    enc_kernel <<<NNODES, 128, 0, stream>>>(x, enc_w1, enc_b1, enc_w2, enc_b2, enc);
    proj_kernel<<<NNODES, 128, 0, stream>>>(enc, ipw, ipb, pw1, cw1, q, k, sap, sbp, sac, sbc);
    attn_kernel<<<NNODES, 128, 0, stream>>>(q, k, out_attn);
    edge_kernel<<<NNODES, 256, 0, stream>>>(sap, sbp, sac, sbc, pb1, pw2, pb2,
                                            cb1, cw2, cb2, out_prob, out_ef);
}
